// transformer_layer_76940044140882
// MI455X (gfx1250) — compile-verified, matrix operand order checked by reading, not run on hardware
//
#include <hip/hip_runtime.h>

// ---------------------------------------------------------------------------
// Types for CDNA5 WMMA (wave32, 16x16x32 bf16 -> f32 accumulate)
// ---------------------------------------------------------------------------
typedef __attribute__((ext_vector_type(16))) __bf16 bf16x16;
typedef __attribute__((ext_vector_type(8)))  __bf16 bf16x8;
typedef __attribute__((ext_vector_type(8)))  float  f32x8;

#define D_MODEL 1024
#define SEQ     2048
#define BATCH   4
#define N_HEAD  16
#define N_HID   64
#define ROWS    (BATCH * SEQ)          // 8192
#define LDS_PITCH 40                   // halves; 80B rows -> 16B aligned b128 loads

// Load a 16-lane-row fragment in the ISA 16-bit A layout from an LDS row:
// lane L (L&15 = matrix row within tile), kgroup = (L>>4)*8.
// halves[0..7]  = K = kgroup..kgroup+7
// halves[8..15] = K = 16+kgroup..16+kgroup+7
__device__ __forceinline__ bf16x16 load_frag(const __bf16* rowp, int kg) {
    const bf16x8 lo = *(const bf16x8*)(rowp + kg);
    const bf16x8 hi = *(const bf16x8*)(rowp + 16 + kg);
    bf16x16 f;
#pragma unroll
    for (int i = 0; i < 8; ++i) { f[i] = lo[i]; f[8 + i] = hi[i]; }
    return f;
}

// ---------------------------------------------------------------------------
// Generic GEMM: C[M,N] = act(A[M,K] @ B[K,N] + bias)
// fp32 in/out, bf16 WMMA compute. EPI (compile-time): 0=none, 1=+bias, 2=+bias,ReLU
// Block: 128x128 tile, 256 threads = 8 waves (4x2), wave tile 32x64 (2x4 WMMA)
// Double-buffered LDS + register prefetch: global loads for tile k+1 are in
// flight while the 8 WMMAs for tile k execute.
// M,N,K must be multiples of 128/128/32 (true for all uses here).
// ---------------------------------------------------------------------------
template <int EPI>
__global__ void __launch_bounds__(256)
tl_gemm_bf16(const float* __restrict__ A, const float* __restrict__ B,
             const float* __restrict__ bias, float* __restrict__ C,
             int N, int Kd) {
    __shared__ __bf16 sA[2][128][LDS_PITCH];   // A tile, row-major [m][k]
    __shared__ __bf16 sB[2][128][LDS_PITCH];   // B^T tile, [n][k]

    const int tid   = threadIdx.x;
    const int lane  = tid & 31;
    const int wave  = tid >> 5;
    const int waveM = wave & 3;             // 4 row groups of 32
    const int waveN = wave >> 2;            // 2 col groups of 64
    const int m0 = blockIdx.x * 128;
    const int n0 = blockIdx.y * 128;

    const int aC4 = (tid & 7) * 4;          // A stage: col within 32-k tile
    const int aR  = tid >> 3;               // A stage: row group base
    const int bN4 = (tid & 31) * 4;         // B stage: n within 128-n tile
    const int bK  = tid >> 5;               // B stage: k group base

    f32x8  acc[2][4] = {};
    float4 pa[4], pb[4];

    auto gload = [&](int k0) {
#pragma unroll
        for (int p = 0; p < 4; ++p) {
            pa[p] = *(const float4*)&A[(size_t)(m0 + p * 32 + aR) * Kd + k0 + aC4];
            pb[p] = *(const float4*)&B[(size_t)(k0 + p * 8 + bK) * N + n0 + bN4];
        }
    };
    auto lstore = [&](int buf) {
#pragma unroll
        for (int p = 0; p < 4; ++p) {
            const int r = p * 32 + aR;
            sA[buf][r][aC4 + 0] = (__bf16)pa[p].x;
            sA[buf][r][aC4 + 1] = (__bf16)pa[p].y;
            sA[buf][r][aC4 + 2] = (__bf16)pa[p].z;
            sA[buf][r][aC4 + 3] = (__bf16)pa[p].w;
            const int kk = p * 8 + bK;
            sB[buf][bN4 + 0][kk] = (__bf16)pb[p].x;
            sB[buf][bN4 + 1][kk] = (__bf16)pb[p].y;
            sB[buf][bN4 + 2][kk] = (__bf16)pb[p].z;
            sB[buf][bN4 + 3][kk] = (__bf16)pb[p].w;
        }
    };

    const int niter = Kd >> 5;              // Kd / 32
    gload(0);
    lstore(0);

    const int kg = (lane >> 4) * 8;
    const int ar = lane & 15;

    for (int it = 0; it < niter; ++it) {
        __syncthreads();
        const int buf = it & 1;
        if (it + 1 < niter) gload((it + 1) << 5);   // prefetch next tile

        bf16x16 afrag[2], bfrag[4];
#pragma unroll
        for (int i = 0; i < 2; ++i)
            afrag[i] = load_frag(&sA[buf][waveM * 32 + i * 16 + ar][0], kg);
#pragma unroll
        for (int j = 0; j < 4; ++j)
            bfrag[j] = load_frag(&sB[buf][waveN * 64 + j * 16 + ar][0], kg);

#pragma unroll
        for (int i = 0; i < 2; ++i)
#pragma unroll
            for (int j = 0; j < 4; ++j)
                acc[i][j] = __builtin_amdgcn_wmma_f32_16x16x32_bf16(
                    false, afrag[i], false, bfrag[j], (short)0, acc[i][j],
                    false, false);

        if (it + 1 < niter) lstore(1 - buf);        // fill other buffer
    }

    // --- epilogue: D layout (VGPR r, lane L): row = r + 8*(L>>4), col = L&15
    const int rr = 8 * (lane >> 4);
    const int cc = lane & 15;
#pragma unroll
    for (int j = 0; j < 4; ++j) {
        const int gcol = n0 + waveN * 64 + j * 16 + cc;
        const float bv = (EPI >= 1) ? bias[gcol] : 0.f;
#pragma unroll
        for (int i = 0; i < 2; ++i)
#pragma unroll
            for (int r = 0; r < 8; ++r) {
                const int grow = m0 + waveM * 32 + i * 16 + r + rr;
                float v = acc[i][j][r] + bv;
                if (EPI == 2) v = fmaxf(v, 0.f);
                C[(size_t)grow * N + gcol] = v;
            }
    }
}

// ---------------------------------------------------------------------------
// Per (b,h): M = K_h^T @ V_h  (64x64, inner dim S=2048) via WMMA.
// One block per (b,h), 128 threads = 4 waves; wave w owns output rows 16w..16w+15.
// Double-buffered LDS + register prefetch over the 64 k-steps.
// ---------------------------------------------------------------------------
__global__ void __launch_bounds__(128)
tl_ktv(const float* __restrict__ K, const float* __restrict__ V,
       float* __restrict__ Mout) {
    __shared__ __bf16 sK[2][64][LDS_PITCH];    // K^T chunk: [d][t]
    __shared__ __bf16 sV[2][64][LDS_PITCH];    // V^T chunk: [d][t]

    const int bh = blockIdx.x;
    const int b = bh >> 4, h = bh & 15;
    const int tid = threadIdx.x, lane = tid & 31, wave = tid >> 5;
    const size_t base = ((size_t)b * SEQ) * D_MODEL + h * N_HID;

    const int d4 = (tid & 15) * 4;
    const int tR = tid >> 4;

    f32x8  acc[4] = {};
    float4 pk[4], pv[4];

    auto gload = [&](int t0) {
#pragma unroll
        for (int p = 0; p < 4; ++p) {
            const size_t off = base + (size_t)(t0 + p * 8 + tR) * D_MODEL + d4;
            pk[p] = *(const float4*)&K[off];
            pv[p] = *(const float4*)&V[off];
        }
    };
    auto lstore = [&](int buf) {
#pragma unroll
        for (int p = 0; p < 4; ++p) {
            const int t = p * 8 + tR;
            sK[buf][d4 + 0][t] = (__bf16)pk[p].x;
            sK[buf][d4 + 1][t] = (__bf16)pk[p].y;
            sK[buf][d4 + 2][t] = (__bf16)pk[p].z;
            sK[buf][d4 + 3][t] = (__bf16)pk[p].w;
            sV[buf][d4 + 0][t] = (__bf16)pv[p].x;
            sV[buf][d4 + 1][t] = (__bf16)pv[p].y;
            sV[buf][d4 + 2][t] = (__bf16)pv[p].z;
            sV[buf][d4 + 3][t] = (__bf16)pv[p].w;
        }
    };

    const int niter = SEQ >> 5;             // 64
    gload(0);
    lstore(0);

    const int kg = (lane >> 4) * 8;
    const int ar = lane & 15;

    for (int it = 0; it < niter; ++it) {
        __syncthreads();
        const int buf = it & 1;
        if (it + 1 < niter) gload((it + 1) << 5);

        const bf16x16 afrag = load_frag(&sK[buf][wave * 16 + ar][0], kg);
#pragma unroll
        for (int j = 0; j < 4; ++j) {
            const bf16x16 bfrag = load_frag(&sV[buf][j * 16 + ar][0], kg);
            acc[j] = __builtin_amdgcn_wmma_f32_16x16x32_bf16(
                false, afrag, false, bfrag, (short)0, acc[j], false, false);
        }

        if (it + 1 < niter) lstore(1 - buf);
    }

    const int rr = 8 * (lane >> 4);
    const int cc = lane & 15;
#pragma unroll
    for (int j = 0; j < 4; ++j)
#pragma unroll
        for (int r = 0; r < 8; ++r)
            Mout[((size_t)bh * 64 + wave * 16 + r + rr) * 64 + j * 16 + cc] =
                acc[j][r];
}

// ---------------------------------------------------------------------------
// ksum[b,h,d] = sum_t K[b,t,h*64+d]
// ---------------------------------------------------------------------------
__global__ void __launch_bounds__(64)
tl_ksum(const float* __restrict__ K, float* __restrict__ ksum) {
    const int bh = blockIdx.x;
    const int b = bh >> 4, h = bh & 15;
    const int d = threadIdx.x;
    const float* base = K + ((size_t)b * SEQ) * D_MODEL + h * N_HID + d;
    float s = 0.f;
    for (int t = 0; t < SEQ; ++t) s += base[(size_t)t * D_MODEL];
    ksum[bh * 64 + d] = s;
}

// ---------------------------------------------------------------------------
// attn[row, d] = (q_h . M_h[:,d]) / 8  /  ((q_h . ksum_h)/8 + 1e-9)
// (linear-attention closed form of the sum-normalized attention)
// ---------------------------------------------------------------------------
__global__ void __launch_bounds__(256)
tl_attn_apply(const float* __restrict__ Q, const float* __restrict__ Mh,
              const float* __restrict__ ksum, float* __restrict__ attn) {
    const int row = blockIdx.x;                 // 0..8191
    const int b = row >> 11;                    // row / 2048
    __shared__ float q[D_MODEL];
    for (int i = threadIdx.x; i < D_MODEL; i += 256)
        q[i] = Q[(size_t)row * D_MODEL + i];
    __syncthreads();

    const int d0 = threadIdx.x * 4;             // 4 consecutive dims, same head
    const int h  = d0 >> 6;
    const int c0 = d0 & 63;
    const float* qh = &q[h * 64];
    const float* Mp = &Mh[(size_t)(b * 16 + h) * 64 * 64];
    const float* ks = &ksum[(b * 16 + h) * 64];

    float den = 0.f;
    float num0 = 0.f, num1 = 0.f, num2 = 0.f, num3 = 0.f;
#pragma unroll 8
    for (int j = 0; j < 64; ++j) {
        const float qv = qh[j];
        den += qv * ks[j];
        const float* mrow = &Mp[j * 64 + c0];
        num0 += qv * mrow[0];
        num1 += qv * mrow[1];
        num2 += qv * mrow[2];
        num3 += qv * mrow[3];
    }
    const float inv = 0.125f / (den * 0.125f + 1e-9f);
    float* o = &attn[(size_t)row * D_MODEL + d0];
    o[0] = num0 * inv; o[1] = num1 * inv; o[2] = num2 * inv; o[3] = num3 * inv;
}

// ---------------------------------------------------------------------------
// out = LayerNorm(a + res) * g + b, optionally masked by t < seq_len[b]
// one block per row, 256 threads, shared-memory reduction
// ---------------------------------------------------------------------------
template <int USE_MASK>
__global__ void __launch_bounds__(256)
tl_add_ln(const float* __restrict__ a, const float* __restrict__ res,
          const float* __restrict__ g, const float* __restrict__ bb,
          float* __restrict__ out, const int* __restrict__ seq_len) {
    const int row = blockIdx.x;
    const int tid = threadIdx.x;
    __shared__ float vals[D_MODEL];
    __shared__ float red[256];
    __shared__ float s_mu, s_rstd;

    float s = 0.f;
#pragma unroll
    for (int i = 0; i < 4; ++i) {
        const int idx = tid + i * 256;
        const float v = a[(size_t)row * D_MODEL + idx] +
                        res[(size_t)row * D_MODEL + idx];
        vals[idx] = v; s += v;
    }
    red[tid] = s; __syncthreads();
    for (int off = 128; off > 0; off >>= 1) {
        if (tid < off) red[tid] += red[tid + off];
        __syncthreads();
    }
    if (tid == 0) s_mu = red[0] * (1.f / (float)D_MODEL);
    __syncthreads();
    const float mu = s_mu;

    float sq = 0.f;
#pragma unroll
    for (int i = 0; i < 4; ++i) {
        const int idx = tid + i * 256;
        const float d = vals[idx] - mu;
        sq += d * d;
    }
    red[tid] = sq; __syncthreads();
    for (int off = 128; off > 0; off >>= 1) {
        if (tid < off) red[tid] += red[tid + off];
        __syncthreads();
    }
    if (tid == 0) s_rstd = rsqrtf(red[0] * (1.f / (float)D_MODEL) + 1e-5f);
    __syncthreads();
    const float rstd = s_rstd;

    float m = 1.f;
    if (USE_MASK) {
        const int b = row >> 11;
        const int t = row & (SEQ - 1);
        m = (t < seq_len[b]) ? 1.f : 0.f;
    }
#pragma unroll
    for (int i = 0; i < 4; ++i) {
        const int idx = tid + i * 256;
        out[(size_t)row * D_MODEL + idx] =
            ((vals[idx] - mu) * rstd * g[idx] + bb[idx]) * m;
    }
}

// ---------------------------------------------------------------------------
// Host launch
// ---------------------------------------------------------------------------
extern "C" void kernel_launch(void* const* d_in, const int* in_sizes, int n_in,
                              void* d_out, int out_size, void* d_ws, size_t ws_size,
                              hipStream_t stream) {
    (void)in_sizes; (void)n_in; (void)out_size; (void)ws_size;
    const float* x     = (const float*)d_in[0];
    const int*   seqL  = (const int*)d_in[1];
    const float* Wq    = (const float*)d_in[2];
    const float* Wk    = (const float*)d_in[3];
    const float* Wv    = (const float*)d_in[4];
    const float* Wg    = (const float*)d_in[5];
    const float* ln1g  = (const float*)d_in[6];
    const float* ln1b  = (const float*)d_in[7];
    const float* W1    = (const float*)d_in[8];
    const float* b1    = (const float*)d_in[9];
    const float* W2    = (const float*)d_in[10];
    const float* b2    = (const float*)d_in[11];
    const float* ln2g  = (const float*)d_in[12];
    const float* ln2b  = (const float*)d_in[13];
    float* out = (float*)d_out;

    const size_t TSZ = (size_t)ROWS * D_MODEL;          // 8M floats = 32MB
    float* buf0 = (float*)d_ws;                          // Q -> ff1
    float* buf1 = buf0 + TSZ;                            // K -> y
    float* buf2 = buf1 + TSZ;                            // V -> h
    float* buf3 = buf2 + TSZ;                            // attn -> ff2
    float* Mbuf = buf3 + TSZ;                            // 64 * 64*64 = 256K floats
    float* ksum = Mbuf + (size_t)64 * 64 * 64;           // 64 * 64 floats

    const dim3 gg(ROWS / 128, D_MODEL / 128);            // 64 x 8

    // QKV projections
    tl_gemm_bf16<0><<<gg, 256, 0, stream>>>(x, Wq, nullptr, buf0, D_MODEL, D_MODEL);
    tl_gemm_bf16<0><<<gg, 256, 0, stream>>>(x, Wk, nullptr, buf1, D_MODEL, D_MODEL);
    tl_gemm_bf16<0><<<gg, 256, 0, stream>>>(x, Wv, nullptr, buf2, D_MODEL, D_MODEL);

    // linear-attention reductions: ksum and K^T V per (b,h)
    tl_ksum<<<BATCH * N_HEAD, 64, 0, stream>>>(buf1, ksum);
    tl_ktv<<<BATCH * N_HEAD, 128, 0, stream>>>(buf1, buf2, Mbuf);

    // attn = (Q @ KtV) / (Q @ ksum + eps)
    tl_attn_apply<<<ROWS, 256, 0, stream>>>(buf0, Mbuf, ksum, buf3);

    // output projection, residual + LN1
    tl_gemm_bf16<0><<<gg, 256, 0, stream>>>(buf3, Wg, nullptr, buf1, D_MODEL, D_MODEL);
    tl_add_ln<0><<<ROWS, 256, 0, stream>>>(buf1, x, ln1g, ln1b, buf2, nullptr);

    // FFN: relu(h@W1+b1) @ W2 + b2, residual + LN2 + length mask
    tl_gemm_bf16<2><<<gg, 256, 0, stream>>>(buf2, W1, b1, buf0, D_MODEL, D_MODEL);
    tl_gemm_bf16<1><<<gg, 256, 0, stream>>>(buf0, W2, b2, buf3, D_MODEL, D_MODEL);
    tl_add_ln<1><<<ROWS, 256, 0, stream>>>(buf3, buf2, ln2g, ln2b, out, seqL);
}
